// HydraPool_85976655331857
// MI455X (gfx1250) — matrix operand order repaired and checked by reading, not compiled, not run on hardware
//
#include <hip/hip_runtime.h>
#include <hip/hip_bf16.h>

// ---------------------------------------------------------------------------
// HydraPool forward for MI455X (gfx1250, wave32, WMMA bf16 path)
//   B=4 S=256 D_IN=1024 ATTN=1024 HD=64 H=16 C=4096 R=256 CR=1024 SC=2048
//   FF=3072 EPS=1e-5
// All WMMA fragment loads are contiguous vector loads (b128s); FF GEMMs stage
// B tiles through LDS with gfx1250 async global->LDS copies when available.
// ---------------------------------------------------------------------------

typedef __attribute__((ext_vector_type(16))) __bf16 v16bf;
typedef __attribute__((ext_vector_type(8)))  __bf16 v8bf;
typedef __attribute__((ext_vector_type(4)))  __bf16 v4bf;
typedef __attribute__((ext_vector_type(8)))  float  v8f;
typedef __attribute__((ext_vector_type(4)))  float  v4f;
typedef __attribute__((__vector_size__(16))) int    v4i;  // async-lds payload

#define DEV __device__ __forceinline__

#if defined(__has_builtin)
#  if __has_builtin(__builtin_amdgcn_global_load_async_to_lds_b128)
#    define ASYNC_LDS 1
#  else
#    define ASYNC_LDS 0
#  endif
#  if __has_builtin(__builtin_amdgcn_s_wait_asynccnt)
#    define HAVE_WAIT_ASYNC 1
#  else
#    define HAVE_WAIT_ASYNC 0
#  endif
#else
#  define ASYNC_LDS 0
#  define HAVE_WAIT_ASYNC 0
#endif

// 16 bytes global -> LDS (async when the toolchain exposes the builtin)
DEV void stage_b128(const __bf16* g, __bf16* l) {
#if ASYNC_LDS
  __builtin_amdgcn_global_load_async_to_lds_b128(
      (__attribute__((address_space(1))) v4i*)(g),
      (__attribute__((address_space(3))) v4i*)(l), 0, 0);
#else
  *(v8bf*)l = *(const v8bf*)g;
#endif
}
DEV void async_wait0() {
#if ASYNC_LDS
#  if HAVE_WAIT_ASYNC
  __builtin_amdgcn_s_wait_asynccnt(0);
#  else
  asm volatile("s_wait_asynccnt 0x0" ::: "memory");
#  endif
#endif
}

DEV v8f wmma_bf16(v16bf a, v16bf b, v8f c) {
  return __builtin_amdgcn_wmma_f32_16x16x32_bf16(
      false, a, false, b, (short)0, c, false, false);
}

// A fragment (16x32 tile of row-major MxK): per lane two contiguous 16B loads.
// lanes 0-15: row m0+l, elems 0..7 -> K k0+0..7, 8..15 -> K k0+16..23;
// lanes 16-31: K k0+8..15 / k0+24..31.
DEV v16bf load_a_frag(const __bf16* A, int lda, int m0, int k0, int lane) {
  int l = lane & 15, hi = lane >> 4;
  const __bf16* row = A + (long)(m0 + l) * lda + k0 + hi * 8;
  v8bf lo = *(const v8bf*)(row);
  v8bf hh = *(const v8bf*)(row + 16);
  v16bf f;
#pragma unroll
  for (int e = 0; e < 8; ++e) { f[e] = lo[e]; f[e + 8] = hh[e]; }
  return f;
}

// B fragment (32x16 tile at (k0, n0)) where element (k,n) = B[n*ldb + k]
// (K contiguous): per lane two contiguous 16B loads.
// lane = column n0+(lane&15); lanes 0-15 K k0+0..7 / k0+16..23, lanes 16-31 K k0+8..15 / k0+24..31.
DEV v16bf load_b_contig(const __bf16* B, long ldb, int k0, int n0, int lane) {
  int l = lane & 15, hi = lane >> 4;
  const __bf16* col = B + (long)(n0 + l) * ldb + k0 + hi * 8;
  v8bf lo = *(const v8bf*)(col);
  v8bf hh = *(const v8bf*)(col + 16);
  v16bf f;
#pragma unroll
  for (int e = 0; e < 8; ++e) { f[e] = lo[e]; f[e + 8] = hh[e]; }
  return f;
}

DEV float wave_sum(float v) {
#pragma unroll
  for (int o = 16; o > 0; o >>= 1) v += __shfl_xor(v, o, 32);
  return v;
}
DEV float wave_max(float v) {
#pragma unroll
  for (int o = 16; o > 0; o >>= 1) v = fmaxf(v, __shfl_xor(v, o, 32));
  return v;
}
DEV float siluf(float x) { return x / (1.f + __expf(-x)); }

// ---------------------------------------------------------------------------
// Small prep kernels
// ---------------------------------------------------------------------------

__global__ __launch_bounds__(256) void rmsnorm64_f32(const float* __restrict__ in,
                                                     float* __restrict__ out,
                                                     int nrows) {
  int row = blockIdx.x * 8 + (threadIdx.x >> 5);
  int lane = threadIdx.x & 31;
  if (row >= nrows) return;
  const float* r = in + (long)row * 64;
  float a = r[lane], b = r[lane + 32];
  float ss = wave_sum(a * a + b * b);
  float sc = rsqrtf(ss * (1.f / 64.f) + 1e-5f);
  out[(long)row * 64 + lane] = a * sc;
  out[(long)row * 64 + lane + 32] = b * sc;
}

__global__ __launch_bounds__(256) void rmsnorm64_bf16(const float* __restrict__ in,
                                                      __bf16* __restrict__ out,
                                                      int nrows) {
  int row = blockIdx.x * 8 + (threadIdx.x >> 5);
  int lane = threadIdx.x & 31;
  if (row >= nrows) return;
  const float* r = in + (long)row * 64;
  float a = r[lane], b = r[lane + 32];
  float ss = wave_sum(a * a + b * b);
  float sc = rsqrtf(ss * (1.f / 64.f) + 1e-5f);
  out[(long)row * 64 + lane] = (__bf16)(a * sc);
  out[(long)row * 64 + lane + 32] = (__bf16)(b * sc);
}

__global__ __launch_bounds__(256) void scatter_add64(
    const float* __restrict__ src, float* __restrict__ dst,
    const int* __restrict__ idx_src, const int* __restrict__ idx_dst,
    const float* __restrict__ w, int per_h, int srcC, int dstC) {
  int g = blockIdx.x * 8 + (threadIdx.x >> 5);
  int lane = threadIdx.x & 31;
  if (g >= 16 * per_h) return;
  int h = g / per_h, j = g - h * per_h;
  int si = idx_src[j], di = idx_dst[j];
  float ww = w[(long)h * per_h + j];
  const float* s = src + ((long)h * srcC + si) * 64;
  float* d = dst + ((long)h * dstC + di) * 64;
  atomicAdd(&d[lane], s[lane] * ww);
  atomicAdd(&d[lane + 32], s[lane + 32] * ww);
}

// vectorized f32 -> bf16 (4 elements / thread)
__global__ __launch_bounds__(256) void cvt_f32_bf16x4(const float* __restrict__ in,
                                                      __bf16* __restrict__ out,
                                                      long n4) {
  long i = (long)blockIdx.x * blockDim.x + threadIdx.x;
  if (i >= n4) return;
  v4f v = ((const v4f*)in)[i];
  v4bf o;
#pragma unroll
  for (int e = 0; e < 4; ++e) o[e] = (__bf16)v[e];
  ((v4bf*)out)[i] = o;
}

// ---------------------------------------------------------------------------
// Generic WMMA GEMM (direct from global): C = A(MxK) * B, B[n*ldb+k].
// Wave 32x32 tile; block 8 waves (2x4) -> 64x128.
// ---------------------------------------------------------------------------
__global__ __launch_bounds__(256) void gemm_bf16_f32(
    const __bf16* __restrict__ A, const __bf16* __restrict__ B,
    float* __restrict__ C, int M, int N, int K, int lda, long ldb, int ldc) {
  int lane = threadIdx.x & 31, wave = threadIdx.x >> 5;
  int m0 = blockIdx.x * 64 + (wave >> 2) * 32;
  int n0 = blockIdx.y * 128 + (wave & 3) * 32;
  if (m0 >= M || n0 >= N) return;
  v8f acc[2][2] = {};
  for (int k0 = 0; k0 < K; k0 += 32) {
    __builtin_prefetch(A + (long)m0 * lda + k0 + 128, 0, 0);
    v16bf a0 = load_a_frag(A, lda, m0, k0, lane);
    v16bf a1 = load_a_frag(A, lda, m0 + 16, k0, lane);
    v16bf b0 = load_b_contig(B, ldb, k0, n0, lane);
    v16bf b1 = load_b_contig(B, ldb, k0, n0 + 16, lane);
    acc[0][0] = wmma_bf16(a0, b0, acc[0][0]);
    acc[0][1] = wmma_bf16(a0, b1, acc[0][1]);
    acc[1][0] = wmma_bf16(a1, b0, acc[1][0]);
    acc[1][1] = wmma_bf16(a1, b1, acc[1][1]);
  }
  int l = lane & 15, hi = lane >> 4;
#pragma unroll
  for (int i = 0; i < 2; ++i)
#pragma unroll
    for (int j = 0; j < 2; ++j) {
      int col = n0 + j * 16 + l;
#pragma unroll
      for (int r = 0; r < 8; ++r)
        C[(long)(m0 + i * 16 + r + hi * 8) * ldc + col] = acc[i][j][r];
    }
}

// ---------------------------------------------------------------------------
// kv post: kv f32 (B,S,2,H,HD) -> k_bf16 rmsnormed (B,H,S,HD) and
// v transposed bf16 (B,H,HD,S) so the PV GEMM gets contiguous B loads.
// ---------------------------------------------------------------------------
__global__ __launch_bounds__(256) void kv_post(const float* __restrict__ kv,
                                               __bf16* __restrict__ Kb,
                                               __bf16* __restrict__ Vt) {
  int row = blockIdx.x * 8 + (threadIdx.x >> 5);  // (b, s, h), 16384 rows
  int lane = threadIdx.x & 31;
  int h = row & 15, s = (row >> 4) & 255, b = row >> 12;
  const float* kr = kv + ((long)(b * 256 + s) * 2 + 0) * 1024 + h * 64;
  const float* vr = kv + ((long)(b * 256 + s) * 2 + 1) * 1024 + h * 64;
  float k0 = kr[lane], k1 = kr[lane + 32];
  float ss = wave_sum(k0 * k0 + k1 * k1);
  float sc = rsqrtf(ss * (1.f / 64.f) + 1e-5f);
  long bh = b * 16 + h;
  long ob = (bh * 256 + s) * 64;
  Kb[ob + lane] = (__bf16)(k0 * sc);
  Kb[ob + lane + 32] = (__bf16)(k1 * sc);
  Vt[(bh * 64 + lane) * 256 + s] = (__bf16)vr[lane];
  Vt[(bh * 64 + lane + 32) * 256 + s] = (__bf16)vr[lane + 32];
}

// ---------------------------------------------------------------------------
// Fused attention: per (b,h), per wave 16 rows of c.
//   scores(16x256) -> LDS, softmax in LDS, o = attn @ v (v pre-transposed).
// LDS: 4 waves * 16*256 f32 = 64 KB.
// ---------------------------------------------------------------------------
__global__ __launch_bounds__(128) void attn_fused(const __bf16* __restrict__ Q,
                                                  const __bf16* __restrict__ Kb,
                                                  const __bf16* __restrict__ Vt,
                                                  float* __restrict__ O) {
  __shared__ float sm[4 * 16 * 256];
  int lane = threadIdx.x & 31, wave = threadIdx.x >> 5;
  int bh = blockIdx.y, b = bh >> 4, h = bh & 15;
  int c0 = blockIdx.x * 64 + wave * 16;
  float* srow = sm + wave * 16 * 256;
  const __bf16* qh = Q + (long)h * 4096 * 64;
  const __bf16* kh = Kb + (long)bh * 256 * 64;
  const __bf16* vh = Vt + (long)bh * 64 * 256;
  int l = lane & 15, hi = lane >> 4;

  v16bf aq0 = load_a_frag(qh, 64, c0, 0, lane);
  v16bf aq1 = load_a_frag(qh, 64, c0, 32, lane);

  // scores = q @ k^T * 1/8 -> LDS   (B element (k,n) = kh[n*64+k], ldb=64)
#pragma unroll
  for (int st = 0; st < 16; ++st) {
    v8f acc = {};
    acc = wmma_bf16(aq0, load_b_contig(kh, 64, 0, st * 16, lane), acc);
    acc = wmma_bf16(aq1, load_b_contig(kh, 64, 32, st * 16, lane), acc);
    int col = st * 16 + l;
#pragma unroll
    for (int r = 0; r < 8; ++r) srow[(r + hi * 8) * 256 + col] = acc[r] * 0.125f;
  }

  // softmax (wave-private rows; lanes cooperate per row)
  for (int r = 0; r < 16; ++r) {
    float* rp = srow + r * 256;
    float m = -3.0e38f;
    for (int j = lane; j < 256; j += 32) m = fmaxf(m, rp[j]);
    m = wave_max(m);
    float s = 0.f;
    for (int j = lane; j < 256; j += 32) {
      float e = __expf(rp[j] - m);
      rp[j] = e;
      s += e;
    }
    s = wave_sum(s);
    float inv = 1.f / s;
    for (int j = lane; j < 256; j += 32) rp[j] *= inv;
  }

  // o = attn @ v   (B element (k=s, n=e) = vh[n*256+k], ldb=256)
#pragma unroll
  for (int nt = 0; nt < 4; ++nt) {
    v8f acc = {};
    for (int kk = 0; kk < 8; ++kk) {
      int kb = kk * 32;
      const float* arow = srow + (long)l * 256 + kb + hi * 8;
      v8f a0 = *(const v8f*)(arow);
      v8f a1 = *(const v8f*)(arow + 16);
      v16bf fa;
#pragma unroll
      for (int e = 0; e < 8; ++e) {
        fa[e] = (__bf16)a0[e];
        fa[e + 8] = (__bf16)a1[e];
      }
      acc = wmma_bf16(fa, load_b_contig(vh, 256, kb, nt * 16, lane), acc);
    }
    int col = nt * 16 + l;
#pragma unroll
    for (int r = 0; r < 8; ++r) {
      int c = c0 + r + hi * 8;
      O[(long)(b * 4096 + c) * 1024 + h * 64 + col] = acc[r];
    }
  }
}

// ---------------------------------------------------------------------------
// LayerNorm rows of 1024: o f32 -> f bf16
// ---------------------------------------------------------------------------
__global__ __launch_bounds__(256) void layernorm1024(
    const float* __restrict__ O, const float* __restrict__ gamma,
    const float* __restrict__ beta, __bf16* __restrict__ F) {
  __shared__ float red[256];
  int row = blockIdx.x, t = threadIdx.x;
  const float* r = O + (long)row * 1024;
  float v0 = r[t], v1 = r[t + 256], v2 = r[t + 512], v3 = r[t + 768];
  red[t] = v0 + v1 + v2 + v3;
  __syncthreads();
  for (int o = 128; o > 0; o >>= 1) {
    if (t < o) red[t] += red[t + o];
    __syncthreads();
  }
  float mu = red[0] * (1.f / 1024.f);
  __syncthreads();
  float d0 = v0 - mu, d1 = v1 - mu, d2 = v2 - mu, d3 = v3 - mu;
  red[t] = d0 * d0 + d1 * d1 + d2 * d2 + d3 * d3;
  __syncthreads();
  for (int o = 128; o > 0; o >>= 1) {
    if (t < o) red[t] += red[t + o];
    __syncthreads();
  }
  float inv = rsqrtf(red[0] * (1.f / 1024.f) + 1e-5f);
  __bf16* fr = F + (long)row * 1024;
  fr[t]       = (__bf16)(d0 * inv * gamma[t] + beta[t]);
  fr[t + 256] = (__bf16)(d1 * inv * gamma[t + 256] + beta[t + 256]);
  fr[t + 512] = (__bf16)(d2 * inv * gamma[t + 512] + beta[t + 512]);
  fr[t + 768] = (__bf16)(d3 * inv * gamma[t + 768] + beta[t + 768]);
}

// ---------------------------------------------------------------------------
// FF-in GEMM + SwiGLU.  M=16384, K=1024; per block: cols [n,n+128) of f1 and
// the matching f2 cols (+3072).  B tiles staged in LDS via async copies.
// LDS: 256 rows x 32 halves = 16 KB.
// ---------------------------------------------------------------------------
__global__ __launch_bounds__(256) void gemm_ffin_swiglu(
    const __bf16* __restrict__ A, const __bf16* __restrict__ Bw,
    __bf16* __restrict__ G) {
  __shared__ __bf16 btile[256 * 32];
  const int K = 1024;
  int lane = threadIdx.x & 31, wave = threadIdx.x >> 5;
  int m0 = blockIdx.x * 64 + (wave >> 2) * 32;
  int nl = (wave & 3) * 32;                 // local col base within 128
  int nblk = blockIdx.y * 128;              // global f1 col base
  int srow = threadIdx.x;                   // staging row 0..255
  const __bf16* gsrc0 =
      Bw + (long)(nblk + (srow & 127) + (srow >> 7) * 3072) * K;
  __bf16* ldst = btile + srow * 32;
  v8f acc1[2][2] = {}, acc2[2][2] = {};
  for (int k0 = 0; k0 < K; k0 += 32) {
    const __bf16* gsrc = gsrc0 + k0;
    stage_b128(gsrc, ldst);
    stage_b128(gsrc + 8, ldst + 8);
    stage_b128(gsrc + 16, ldst + 16);
    stage_b128(gsrc + 24, ldst + 24);
    async_wait0();
    __syncthreads();
    __builtin_prefetch(A + (long)m0 * K + k0 + 128, 0, 0);
    v16bf a0 = load_a_frag(A, K, m0, k0, lane);
    v16bf a1 = load_a_frag(A, K, m0 + 16, k0, lane);
#pragma unroll
    for (int j = 0; j < 2; ++j) {
      v16bf b1 = load_b_contig(btile, 32, 0, nl + j * 16, lane);
      v16bf b2 = load_b_contig(btile, 32, 0, 128 + nl + j * 16, lane);
      acc1[0][j] = wmma_bf16(a0, b1, acc1[0][j]);
      acc1[1][j] = wmma_bf16(a1, b1, acc1[1][j]);
      acc2[0][j] = wmma_bf16(a0, b2, acc2[0][j]);
      acc2[1][j] = wmma_bf16(a1, b2, acc2[1][j]);
    }
    __syncthreads();
  }
  int l = lane & 15, hi = lane >> 4;
#pragma unroll
  for (int i = 0; i < 2; ++i)
#pragma unroll
    for (int j = 0; j < 2; ++j) {
      int col = nblk + nl + j * 16 + l;
#pragma unroll
      for (int r = 0; r < 8; ++r) {
        int row = m0 + i * 16 + r + hi * 8;
        float f1 = acc1[i][j][r], f2 = acc2[i][j][r];
        G[(long)row * 3072 + col] = (__bf16)(siluf(f1) * f2);
      }
    }
}

// ---------------------------------------------------------------------------
// FF-out GEMM + residual: z = g @ Wout^T + o -> bf16.  M=16384,N=1024,K=3072.
// B tile (128 x 32) staged in LDS via async copies (8 KB).
// ---------------------------------------------------------------------------
__global__ __launch_bounds__(256) void gemm_ffout_res(
    const __bf16* __restrict__ G, const __bf16* __restrict__ Bw,
    const float* __restrict__ O, __bf16* __restrict__ Z) {
  __shared__ __bf16 btile[128 * 32];
  const int K = 3072;
  int lane = threadIdx.x & 31, wave = threadIdx.x >> 5;
  int m0 = blockIdx.x * 64 + (wave >> 2) * 32;
  int nl = (wave & 3) * 32;
  int nblk = blockIdx.y * 128;
  int srow = threadIdx.x >> 1, sseg = (threadIdx.x & 1) * 16;
  const __bf16* gsrc0 = Bw + (long)(nblk + srow) * K + sseg;
  __bf16* ldst = btile + srow * 32 + sseg;
  v8f acc[2][2] = {};
  for (int k0 = 0; k0 < K; k0 += 32) {
    const __bf16* gsrc = gsrc0 + k0;
    stage_b128(gsrc, ldst);
    stage_b128(gsrc + 8, ldst + 8);
    async_wait0();
    __syncthreads();
    __builtin_prefetch(G + (long)m0 * K + k0 + 128, 0, 0);
    v16bf a0 = load_a_frag(G, K, m0, k0, lane);
    v16bf a1 = load_a_frag(G, K, m0 + 16, k0, lane);
    v16bf b0 = load_b_contig(btile, 32, 0, nl, lane);
    v16bf b1 = load_b_contig(btile, 32, 0, nl + 16, lane);
    acc[0][0] = wmma_bf16(a0, b0, acc[0][0]);
    acc[0][1] = wmma_bf16(a0, b1, acc[0][1]);
    acc[1][0] = wmma_bf16(a1, b0, acc[1][0]);
    acc[1][1] = wmma_bf16(a1, b1, acc[1][1]);
    __syncthreads();
  }
  int l = lane & 15, hi = lane >> 4;
#pragma unroll
  for (int i = 0; i < 2; ++i)
#pragma unroll
    for (int j = 0; j < 2; ++j) {
      int col = nblk + nl + j * 16 + l;
#pragma unroll
      for (int r = 0; r < 8; ++r) {
        int row = m0 + i * 16 + r + hi * 8;
        Z[(long)row * 1024 + col] =
            (__bf16)(acc[i][j][r] + O[(long)row * 1024 + col]);
      }
    }
}

// ---------------------------------------------------------------------------
// Output projection: per c, y[b][o] = z[b][c][:] . out_w[c][:][o];
// out = silu(y0)*y1.
// ---------------------------------------------------------------------------
__global__ __launch_bounds__(256) void out_proj(const __bf16* __restrict__ Z,
                                                const float* __restrict__ W,
                                                float* __restrict__ out) {
  __shared__ float sh[8];
  int c = blockIdx.x;
  int lane = threadIdx.x & 31, wave = threadIdx.x >> 5;
  int b = wave >> 1, oidx = wave & 1;
  const __bf16* z = Z + (long)(b * 4096 + c) * 1024;
  const float* w = W + (long)c * 2048 + oidx;
  float s = 0.f;
  for (int a = lane; a < 1024; a += 32) s += (float)z[a] * w[(long)a * 2];
  s = wave_sum(s);
  if (lane == 0) sh[wave] = s;
  __syncthreads();
  if (threadIdx.x < 4) {
    float y0 = sh[threadIdx.x * 2], y1 = sh[threadIdx.x * 2 + 1];
    out[(long)threadIdx.x * 4096 + c] = siluf(y0) * y1;
  }
}

// ---------------------------------------------------------------------------
// Host launcher
// ---------------------------------------------------------------------------
extern "C" void kernel_launch(void* const* d_in, const int* in_sizes, int n_in,
                              void* d_out, int out_size, void* d_ws,
                              size_t ws_size, hipStream_t stream) {
  (void)in_sizes; (void)n_in; (void)out_size; (void)ws_size;
  const float* x       = (const float*)d_in[0];   // (4,256,1024)
  const float* cls     = (const float*)d_in[1];   // (16,4096,64)
  const float* roots   = (const float*)d_in[2];   // (16,256,64)
  const float* crw     = (const float*)d_in[3];   // (16,1024,1)
  const float* ccw     = (const float*)d_in[4];   // (16,2048,1)
  const float* kv_w    = (const float*)d_in[5];   // (2048,1024)
  const float* gamma   = (const float*)d_in[6];   // (1024)
  const float* beta    = (const float*)d_in[7];   // (1024)
  const float* ffin_w  = (const float*)d_in[8];   // (6144,1024)
  const float* ffout_w = (const float*)d_in[9];   // (1024,3072)
  const float* outw    = (const float*)d_in[10];  // (4096,1024,2)
  const int*   cri     = (const int*)d_in[11];    // (2,1024)
  const int*   cci     = (const int*)d_in[12];    // (2,2048)
  float* out = (float*)d_out;                     // (4,4096,1) flat

  char* p = (char*)d_ws;
  float*  q     = (float*)p;  p += (size_t)16 * 4096 * 64 * 4;   // 16 MB
  float*  qdet  = (float*)p;  p += (size_t)16 * 4096 * 64 * 4;   // 16 MB
  float*  qr    = (float*)p;  p += (size_t)16 * 256 * 64 * 4;    //  1 MB
  __bf16* qb    = (__bf16*)p; p += (size_t)16 * 4096 * 64 * 2;   //  8 MB
  __bf16* xb    = (__bf16*)p; p += (size_t)1024 * 1024 * 2;      //  2 MB
  __bf16* kvwb  = (__bf16*)p; p += (size_t)2048 * 1024 * 2;      //  4 MB
  float*  kvf   = (float*)p;  p += (size_t)1024 * 2048 * 4;      //  8 MB
  __bf16* kb    = (__bf16*)p; p += (size_t)1024 * 1024 * 2;      //  2 MB
  __bf16* vt    = (__bf16*)p; p += (size_t)1024 * 1024 * 2;      //  2 MB
  float*  o     = (float*)p;  p += (size_t)16384 * 1024 * 4;     // 64 MB
  __bf16* fb    = (__bf16*)p; p += (size_t)16384 * 1024 * 2;     // 32 MB
  __bf16* ffinb = (__bf16*)p; p += (size_t)6144 * 1024 * 2;      // 12 MB
  __bf16* ffoutb= (__bf16*)p; p += (size_t)1024 * 3072 * 2;      //  6 MB
  __bf16* gb    = (__bf16*)p; p += (size_t)16384 * 3072 * 2;     // 96 MB
  __bf16* zb    = (__bf16*)p; p += (size_t)16384 * 1024 * 2;     // 32 MB

  // --- q construction ---
  (void)hipMemcpyAsync(q, cls, (size_t)16 * 4096 * 64 * 4,
                       hipMemcpyDeviceToDevice, stream);
  rmsnorm64_f32<<<512, 256, 0, stream>>>(roots, qr, 16 * 256);
  scatter_add64<<<2048, 256, 0, stream>>>(qr, q, cri, cri + 1024, crw, 1024,
                                          256, 4096);
  (void)hipMemcpyAsync(qdet, q, (size_t)16 * 4096 * 64 * 4,
                       hipMemcpyDeviceToDevice, stream);
  scatter_add64<<<4096, 256, 0, stream>>>(qdet, q, cci, cci + 2048, ccw, 2048,
                                          4096, 4096);
  rmsnorm64_bf16<<<8192, 256, 0, stream>>>(q, qb, 16 * 4096);

  // --- bf16 conversions ---
  cvt_f32_bf16x4<<<1024, 256, 0, stream>>>(x, xb, 262144L);
  cvt_f32_bf16x4<<<2048, 256, 0, stream>>>(kv_w, kvwb, 524288L);
  cvt_f32_bf16x4<<<6144, 256, 0, stream>>>(ffin_w, ffinb, 1572864L);
  cvt_f32_bf16x4<<<3072, 256, 0, stream>>>(ffout_w, ffoutb, 786432L);

  // --- kv = x @ kv_w^T : M=1024, N=2048, K=1024 ---
  gemm_bf16_f32<<<dim3(16, 16), 256, 0, stream>>>(
      xb, kvwb, kvf, 1024, 2048, 1024, /*lda=*/1024, /*ldb=*/1024L,
      /*ldc=*/2048);
  kv_post<<<2048, 256, 0, stream>>>(kvf, kb, vt);

  // --- fused attention: grid (C/64, B*H) ---
  attn_fused<<<dim3(64, 64), 128, 0, stream>>>(qb, kb, vt, o);

  // --- LayerNorm + FFN (SwiGLU) + residual ---
  layernorm1024<<<16384, 256, 0, stream>>>(o, gamma, beta, fb);
  gemm_ffin_swiglu<<<dim3(256, 24), 256, 0, stream>>>(fb, ffinb, gb);
  gemm_ffout_res<<<dim3(256, 8), 256, 0, stream>>>(gb, ffoutb, o, zb);

  // --- output projection + gated activation ---
  out_proj<<<4096, 256, 0, stream>>>(zb, outw, out);
}
